// Recurrent_LIF_81166291959848
// MI455X (gfx1250) — matrix operand order repaired and checked by reading, not compiled, not run on hardware
//
#include <hip/hip_runtime.h>

typedef __attribute__((ext_vector_type(16))) _Float16 v16h;
typedef __attribute__((ext_vector_type(8)))  _Float16 v8h;
typedef __attribute__((ext_vector_type(8)))  float    v8f;

#define T_STEPS 128
#define B_DIM   128
#define N_DIM   1024
#define NBLK    (N_DIM / 16)   // 64 persistent blocks

// ---------------------------------------------------------------------------
// Init (runs every call): W f32->f16, zero y ping buffer, zero barrier ctr.
// ---------------------------------------------------------------------------
__global__ void lif_init(const float* __restrict__ W,
                         _Float16* __restrict__ Wh,
                         _Float16* __restrict__ y0,
                         unsigned* __restrict__ ctr,
                         int nW, int nState) {
    int i = blockIdx.x * blockDim.x + threadIdx.x;
    if (i < nW)     Wh[i] = (_Float16)W[i];
    if (i < nState) y0[i] = (_Float16)0.0f;
    if (i == 0)     *ctr  = 0u;
}

// ---------------------------------------------------------------------------
// Persistent LIF kernel: 64 blocks x 256 threads (8 waves).
//   block  -> 16-neuron output strip (n0), W strip cached in LDS (32 KB)
//   wave   -> 16-batch-row M tile; v and y_prev for its 8 elems/lane live in
//             registers across all 128 timesteps.
// Per step: 32x v_wmma_f32_16x16x32_f16 (4 independent accumulator chains),
// elementwise LIF update, spike stores, grid-wide barrier.
// ---------------------------------------------------------------------------
__global__ __launch_bounds__(256) void lif_persist(
    const float*    __restrict__ tx,    // [T,B,N]
    const _Float16* __restrict__ Wh,    // [N,N] f16, row-major [n][k]
    const float*    __restrict__ bias,  // [N]
    _Float16*       __restrict__ yA,    // [B,N] spike ping
    _Float16*       __restrict__ yB,    // [B,N] spike pong
    float*          __restrict__ out,   // [T,B,N]
    unsigned*       __restrict__ ctr)   // grid barrier (monotonic)
{
    __shared__ _Float16 wtile[16 * N_DIM];   // 32 KB: rows n0..n0+15 of W

    const int n0 = blockIdx.x * 16;

    // Stage W strip into LDS: 2048 x 16B chunks, 256 threads, 8 iters.
    #pragma unroll
    for (int it = 0; it < 8; ++it) {
        int chunk = threadIdx.x + it * 256;   // 0..2047
        int row   = chunk >> 7;               // /128 chunks per row
        int c8    = chunk & 127;
        *(v8h*)(wtile + row * N_DIM + c8 * 8) =
            *(const v8h*)(Wh + (size_t)(n0 + row) * N_DIM + c8 * 8);
    }
    __syncthreads();

    const int lane = threadIdx.x & 31;
    const int wave = threadIdx.x >> 5;   // 0..7 -> M tile
    const int m0   = wave * 16;
    const int hi   = lane >> 4;
    const int l15  = lane & 15;
    const int gn   = n0 + l15;
    const float bn = bias[gn];

    // B fragment source in LDS: lane holds W row l15, 16 halves at k+hi*16.
    const _Float16* wlds = wtile + l15 * N_DIM + hi * 16;

    // Register-resident LIF state: lane L, slot r <-> (m0 + hi*8 + r, gn).
    float vreg[8], yreg[8];
    #pragma unroll
    for (int r = 0; r < 8; ++r) { vreg[r] = 0.0f; yreg[r] = 0.0f; }

    union AFrag { v16h v; v8h h[2]; };

    for (int t = 0; t < T_STEPS; ++t) {
        const _Float16* yin  = (t & 1) ? yB : yA;
        _Float16*       yout = (t & 1) ? yA : yB;
        const float*    tx_t = tx + (size_t)t * B_DIM * N_DIM;

        // Prefetch next step's tx tile into L2 (global_prefetch_b8).
        if (t + 1 < T_STEPS) {
            const float* txn = tx_t + B_DIM * N_DIM;
            #pragma unroll
            for (int r = 0; r < 8; ++r)
                __builtin_prefetch(txn + (size_t)(m0 + hi * 8 + r) * N_DIM + gn, 0, 1);
        }

        // A fragment source: lane holds y row m0+l15, 16 halves at k+hi*16.
        const _Float16* arow = yin + (size_t)(m0 + l15) * N_DIM + hi * 16;

        v8f acc0 = {}, acc1 = {}, acc2 = {}, acc3 = {};
        #pragma unroll 2
        for (int kk = 0; kk < N_DIM; kk += 128) {
            AFrag a; v16h b;

            a.h[0] = *(const v8h*)(arow + kk);
            a.h[1] = *(const v8h*)(arow + kk + 8);
            b      = *(const v16h*)(wlds + kk);
            acc0 = __builtin_amdgcn_wmma_f32_16x16x32_f16(false, a.v, false, b,
                                                          (short)0, acc0, false, false);
            a.h[0] = *(const v8h*)(arow + kk + 32);
            a.h[1] = *(const v8h*)(arow + kk + 40);
            b      = *(const v16h*)(wlds + kk + 32);
            acc1 = __builtin_amdgcn_wmma_f32_16x16x32_f16(false, a.v, false, b,
                                                          (short)0, acc1, false, false);
            a.h[0] = *(const v8h*)(arow + kk + 64);
            a.h[1] = *(const v8h*)(arow + kk + 72);
            b      = *(const v16h*)(wlds + kk + 64);
            acc2 = __builtin_amdgcn_wmma_f32_16x16x32_f16(false, a.v, false, b,
                                                          (short)0, acc2, false, false);
            a.h[0] = *(const v8h*)(arow + kk + 96);
            a.h[1] = *(const v8h*)(arow + kk + 104);
            b      = *(const v16h*)(wlds + kk + 96);
            acc3 = __builtin_amdgcn_wmma_f32_16x16x32_f16(false, a.v, false, b,
                                                          (short)0, acc3, false, false);
        }
        v8f acc = (acc0 + acc1) + (acc2 + acc3);

        // Elementwise LIF update; v and y_prev stay in registers.
        #pragma unroll
        for (int r = 0; r < 8; ++r) {
            const int gm  = m0 + hi * 8 + r;
            const int idx = gm * N_DIM + gn;
            const float x    = tx_t[idx] + acc[r] + bn;
            const float vnew = 0.5f * vreg[r] * (1.0f - yreg[r]) + x;
            const float sp   = (vnew > 0.5f) ? 1.0f : 0.0f;
            vreg[r] = vnew;
            yreg[r] = sp;
            yout[idx] = (_Float16)sp;
            out[(size_t)t * B_DIM * N_DIM + idx] = sp;
        }

        // Grid-wide barrier (monotonic counter). Release our spike stores,
        // then invalidate WGP$ so next step's y reads are not stale.
        __threadfence();
        __syncthreads();
        if (threadIdx.x == 0) {
            __hip_atomic_fetch_add(ctr, 1u, __ATOMIC_RELEASE, __HIP_MEMORY_SCOPE_AGENT);
            const unsigned target = (unsigned)(t + 1) * (unsigned)NBLK;
            while (__hip_atomic_load(ctr, __ATOMIC_ACQUIRE, __HIP_MEMORY_SCOPE_AGENT) < target)
                __builtin_amdgcn_s_sleep(8);
        }
        __syncthreads();
        __threadfence();
    }
}

// ---------------------------------------------------------------------------
// Workspace: Wh (2 MB) | yA (256 KB) | yB (256 KB) | ctr (4 B)
// ---------------------------------------------------------------------------
extern "C" void kernel_launch(void* const* d_in, const int* in_sizes, int n_in,
                              void* d_out, int out_size, void* d_ws, size_t ws_size,
                              hipStream_t stream) {
    const float* tx   = (const float*)d_in[0];
    const float* W    = (const float*)d_in[1];
    const float* bias = (const float*)d_in[2];
    float* out = (float*)d_out;

    char* ws = (char*)d_ws;
    _Float16* Wh = (_Float16*)ws;
    _Float16* yA = (_Float16*)(ws + (size_t)N_DIM * N_DIM * 2);
    _Float16* yB = yA + (size_t)B_DIM * N_DIM;
    unsigned* ctr = (unsigned*)(yB + (size_t)B_DIM * N_DIM);

    const int nW = N_DIM * N_DIM;
    const int nS = B_DIM * N_DIM;
    lif_init<<<(nW + 255) / 256, 256, 0, stream>>>(W, Wh, yA, ctr, nW, nS);

    lif_persist<<<dim3(NBLK), 256, 0, stream>>>(tx, Wh, bias, yA, yB, out, ctr);
}
